// SparseDecoderMirrorSCA_56530359550000
// MI455X (gfx1250) — matrix operand order repaired and checked by reading, not compiled, not run on hardware
//
#include <hip/hip_runtime.h>
#include <hip/hip_bf16.h>

typedef __attribute__((ext_vector_type(16))) _Float16 v16h;
typedef __attribute__((ext_vector_type(8)))  _Float16 v8h;
typedef __attribute__((ext_vector_type(4)))  _Float16 v4h;
typedef __attribute__((ext_vector_type(8)))  float    v8f;
typedef __attribute__((ext_vector_type(4)))  unsigned int v4u;
typedef __attribute__((ext_vector_type(8)))  int      v8i;
typedef __attribute__((ext_vector_type(4)))  int      v4i;

#define H      4096
#define NB     16
#define BS     256
#define RANK   128
#define TOPK   4
#define ROWS_TILE 64
#define LN_EPS 1e-5f

#define HAVE_TDM __has_builtin(__builtin_amdgcn_tensor_load_to_lds)

// ---------------------------------------------------------------------------
// TDM: build a 2-D tensor D# (data_size = 2 bytes) and issue tensor_load_to_lds.
// Tile (tile_d0 x tile_d1) of a (tensor_d0 x tensor_d1) f16 tensor with row
// stride stride0 (elements) is DMA'd to LDS at lds_off, packed contiguously.
// ---------------------------------------------------------------------------
#if HAVE_TDM
__device__ __forceinline__ void tdm_load_2d_f16(unsigned int lds_off,
                                                const void* gptr,
                                                unsigned int tensor_d0,
                                                unsigned int tensor_d1,
                                                unsigned int tile_d0,
                                                unsigned int tile_d1,
                                                unsigned int stride0)
{
    unsigned long long ga = (unsigned long long)(size_t)gptr;
    v4u g0;
    g0[0] = 1u;                                   // count=1, user descriptor
    g0[1] = lds_off;                              // LDS byte address
    g0[2] = (unsigned int)(ga & 0xffffffffu);     // global addr [95:64]
    g0[3] = (unsigned int)((ga >> 32) & 0x1ffffffu) | (2u << 30); // addr hi | type=2
    v8i g1;
    g1[0] = (int)(1u << 16);                      // data_size=1 (2 bytes), no mask
    g1[1] = (int)((tensor_d0 & 0xffffu) << 16);   // tensor_dim0[15:0]
    g1[2] = (int)(((tensor_d0 >> 16) & 0xffffu) | ((tensor_d1 & 0xffffu) << 16));
    g1[3] = (int)(((tensor_d1 >> 16) & 0xffffu) | ((tile_d0 & 0xffffu) << 16));
    g1[4] = (int)(tile_d1 & 0xffffu);             // tile_dim1, tile_dim2=0
    g1[5] = (int)stride0;                         // tensor_dim0_stride[31:0]
    g1[6] = 0;                                    // stride0 hi | stride1 lo
    g1[7] = 0;
    v4i g2 = {0, 0, 0, 0};
    v4i g3 = {0, 0, 0, 0};
#if defined(__clang_major__) && (__clang_major__ >= 23)
    v8i g4 = {0, 0, 0, 0, 0, 0, 0, 0};
    __builtin_amdgcn_tensor_load_to_lds(g0, g1, g2, g3, g4, 0);
#else
    __builtin_amdgcn_tensor_load_to_lds(g0, g1, g2, g3, 0);
#endif
}
#endif

// ---------------------------------------------------------------------------
// Kernel 0: one-time weight prep. f32 -> f16 and transpose to [n][k] layout:
//   wdt[b][n<128][k<256] = down_w[b][k][n]
//   wut[b][n<256][k<128] = up_w[b][k][n]
// ---------------------------------------------------------------------------
__global__ __launch_bounds__(256)
void SparseSCA_prep_kernel(const float* __restrict__ dw,
                           const float* __restrict__ uw,
                           _Float16* __restrict__ wdt,
                           _Float16* __restrict__ wut)
{
    int idx = blockIdx.x * 256 + threadIdx.x;      // 0 .. NB*BS*RANK-1
    {
        int b = idx >> 15, r = idx & 32767;
        int n = r >> 8, k = r & 255;               // n<128, k<256
        wdt[idx] = (_Float16)dw[(size_t)b * 32768 + (size_t)k * RANK + n];
    }
    {
        int b = idx >> 15, r = idx & 32767;
        int n = r >> 7, k = r & 127;               // n<256, k<128
        wut[idx] = (_Float16)uw[(size_t)b * 32768 + (size_t)k * BS + n];
    }
}

// ---------------------------------------------------------------------------
// Kernel 1: routing/gating. One wave32 per row.
// ---------------------------------------------------------------------------
__global__ __launch_bounds__(256)
void SparseSCA_gate_kernel(const float* __restrict__ x,
                           const float* __restrict__ prior,
                           const float* __restrict__ sw,
                           const float* __restrict__ sb,
                           const float* __restrict__ rps_p,
                           float* __restrict__ gate,
                           int rows)
{
    const int wave = threadIdx.x >> 5;
    const int lane = threadIdx.x & 31;
    const int row  = blockIdx.x * 8 + wave;
    if (row >= rows) return;   // wave-uniform

    const float* xr = x + (size_t)row * H;
    float s = 0.f, s2 = 0.f;
    float d0 = 0.f, d1 = 0.f, d2 = 0.f;
    float w0 = 0.f, w1 = 0.f, w2 = 0.f;
    #pragma unroll 4
    for (int i = lane; i < H; i += 32) {
        float v = xr[i];
        float a = sw[i], b = sw[H + i], c = sw[2 * H + i];
        s += v; s2 += v * v;
        d0 += v * a; d1 += v * b; d2 += v * c;
        w0 += a; w1 += b; w2 += c;
    }
    #pragma unroll
    for (int off = 16; off >= 1; off >>= 1) {
        s  += __shfl_xor(s,  off, 32);
        s2 += __shfl_xor(s2, off, 32);
        d0 += __shfl_xor(d0, off, 32);
        d1 += __shfl_xor(d1, off, 32);
        d2 += __shfl_xor(d2, off, 32);
        w0 += __shfl_xor(w0, off, 32);
        w1 += __shfl_xor(w1, off, 32);
        w2 += __shfl_xor(w2, off, 32);
    }
    if (lane == 0) {
        const float invH = 1.0f / (float)H;
        float mean = s * invH;
        float var  = s2 * invH - mean * mean;
        float rstd = rsqrtf(var + LN_EPS);
        float z0 = rstd * (d0 - mean * w0) + sb[0];
        float z1 = rstd * (d1 - mean * w1) + sb[1];
        float z2 = rstd * (d2 - mean * w2) + sb[2];
        float q0 = 3.0f / (1.0f + expf(-z0));
        float q1 = 3.0f / (1.0f + expf(-z1));
        float q2 = 3.0f / (1.0f + expf(-z2));

        float sp[NB], pr[NB], fs[NB];
        float psum = 0.f;
        #pragma unroll
        for (int j = 0; j < NB; ++j) {
            float cx = (float)(j & 3);
            float cy = (float)((j >> 2) & 3);
            float cz = (float)((j >> 4) & 3);
            float dx = q0 - cx, dy = q1 - cy, dz = q2 - cz;
            sp[j] = expf(-(dx * dx + dy * dy + dz * dz) * 0.5f);
            float pv = prior[(size_t)row * NB + j];
            pv = pv < 0.f ? 0.f : pv;
            pr[j] = pv;
            psum += pv;
        }
        float rps   = rps_p[0];
        float denom = fmaxf(psum, 1e-6f);
        #pragma unroll
        for (int j = 0; j < NB; ++j) {
            float p = pr[j] / denom;
            float b = logf(p + 1e-6f);
            b = fminf(fmaxf(b, -6.0f), 0.0f);
            fs[j] = sp[j] + rps * b;
        }
        int   topi[TOPK];
        float topv[TOPK];
        unsigned used = 0;
        #pragma unroll
        for (int t = 0; t < TOPK; ++t) {
            float best = -1e30f; int bi = 0;
            #pragma unroll
            for (int j = 0; j < NB; ++j) {
                if (!((used >> j) & 1u) && fs[j] > best) { best = fs[j]; bi = j; }
            }
            used |= 1u << bi;
            topi[t] = bi; topv[t] = best;
        }
        float mx = topv[0];
        float ex[TOPK], es = 0.f;
        #pragma unroll
        for (int t = 0; t < TOPK; ++t) { ex[t] = expf(topv[t] - mx); es += ex[t]; }
        float g16[NB];
        #pragma unroll
        for (int j = 0; j < NB; ++j) g16[j] = 0.f;
        #pragma unroll
        for (int t = 0; t < TOPK; ++t) g16[topi[t]] = ex[t] / es;
        #pragma unroll
        for (int j = 0; j < NB; ++j) gate[(size_t)row * NB + j] = g16[j];
    }
}

// ---------------------------------------------------------------------------
// Kernel 2: block-diagonal low-rank adapter.
// grid = (rows/64, NB), 256 threads = 8 waves.
// LDS: sW 64KB (Wd in phase 1, Wu in phase 2, staged by TDM), sA 32KB, sLow 16KB.
// ---------------------------------------------------------------------------
__global__ __launch_bounds__(256)
void SparseSCA_adapter_kernel(const float* __restrict__ x,
                              const _Float16* __restrict__ wdt, // [NB][128][256]
                              const _Float16* __restrict__ wut, // [NB][256][128]
                              const float* __restrict__ gate,
                              const float* __restrict__ res_p,
                              float* __restrict__ out,
                              int rows)
{
    __shared__ _Float16 sW  [RANK * BS];          // 64 KB (Wd then Wu, [n][k])
    __shared__ _Float16 sA  [ROWS_TILE * BS];     // 32 KB, x tile [m][k]
    __shared__ _Float16 sLow[ROWS_TILE * RANK];   // 16 KB, low    [m][k]

    const int row0 = blockIdx.x * ROWS_TILE;
    const int blk  = blockIdx.y;
    const int tid  = threadIdx.x;
    const int lane = tid & 31;
    const int w    = tid >> 5;
    const int nl   = lane & 15;
    const int hi   = lane >> 4;
    const float resScale = res_p[0];

    const float* xblk = x + (size_t)row0 * H + (size_t)blk * BS;
    const _Float16* wdb = wdt + (size_t)blk * BS * RANK;
    const _Float16* wub = wut + (size_t)blk * RANK * BS;

    // ---- stage Wd (TDM, wave 0) overlapped with x tile staging (all waves) --
#if HAVE_TDM
    if (w == 0) {
        tdm_load_2d_f16((unsigned int)(size_t)&sW[0], wdb,
                        /*tensor_d0=*/BS, /*tensor_d1=*/RANK,
                        /*tile_d0=*/BS, /*tile_d1=*/RANK, /*stride0=*/BS);
    }
#else
    #pragma unroll
    for (int j = 0; j < 16; ++j) {
        int q = tid + j * 256;
        *(v8h*)&sW[q * 8] = *(const v8h*)&wdb[q * 8];
    }
#endif
    // x tile: 64x256 f32 -> f16, vectorized
    #pragma unroll
    for (int j = 0; j < 16; ++j) {
        int q  = tid + j * 256;               // float4 index, 4096 total
        int m  = q >> 6;
        int k4 = (q & 63) << 2;
        const float4 v = *(const float4*)&xblk[(size_t)m * H + k4];
        v4h hv;
        hv[0] = (_Float16)v.x; hv[1] = (_Float16)v.y;
        hv[2] = (_Float16)v.z; hv[3] = (_Float16)v.w;
        *(v4h*)&sA[m * BS + k4] = hv;
    }
#if HAVE_TDM
    if (w == 0) __builtin_amdgcn_s_wait_tensorcnt(0);
#endif
    __syncthreads();

    // ---------------- down: low(64x128) = A(64x256) @ Wd(256x128) -----------
    v8f accD[4];
    #pragma unroll
    for (int mt = 0; mt < 4; ++mt)
        #pragma unroll
        for (int e = 0; e < 8; ++e) accD[mt][e] = 0.f;

    #pragma unroll
    for (int kc = 0; kc < 8; ++kc) {
        v16h bf;
        {
            const _Float16* bp = &sW[(w * 16 + nl) * BS + kc * 32];
            #pragma unroll
            for (int e = 0; e < 16; ++e) bf[e] = bp[(e < 8) ? (8 * hi + e) : (16 + 8 * hi + e - 8)];
        }
        #pragma unroll
        for (int mt = 0; mt < 4; ++mt) {
            v16h af;
            const _Float16* ap = &sA[(mt * 16 + nl) * BS + kc * 32];
            #pragma unroll
            for (int e = 0; e < 8; ++e) af[e]     = ap[8 * hi + e];
            #pragma unroll
            for (int e = 0; e < 8; ++e) af[8 + e] = ap[16 + 8 * hi + e];
            accD[mt] = __builtin_amdgcn_wmma_f32_16x16x32_f16(
                false, af, false, bf, (short)0, accD[mt], false, false);
        }
    }

    // spill low -> LDS (f16). D elem r -> M = r + 8*hi, N = nl (tile-local)
    #pragma unroll
    for (int mt = 0; mt < 4; ++mt)
        #pragma unroll
        for (int r = 0; r < 8; ++r) {
            int m = mt * 16 + r + 8 * hi;
            int n = w * 16 + nl;
            sLow[m * RANK + n] = (_Float16)accD[mt][r];
        }
    __syncthreads();   // all Wd reads + low writes complete

    // ---- stage Wu into the same 64KB region --------------------------------
#if HAVE_TDM
    if (w == 0) {
        tdm_load_2d_f16((unsigned int)(size_t)&sW[0], wub,
                        /*tensor_d0=*/RANK, /*tensor_d1=*/BS,
                        /*tile_d0=*/RANK, /*tile_d1=*/BS, /*stride0=*/RANK);
        __builtin_amdgcn_s_wait_tensorcnt(0);
    }
#else
    #pragma unroll
    for (int j = 0; j < 16; ++j) {
        int q = tid + j * 256;
        *(v8h*)&sW[q * 8] = *(const v8h*)&wub[q * 8];
    }
#endif
    __syncthreads();

    // ---------------- up: out_b(64x256) = low(64x128) @ Wu(128x256) ---------
    v8f accU[2][4];
    #pragma unroll
    for (int h = 0; h < 2; ++h)
        #pragma unroll
        for (int mt = 0; mt < 4; ++mt)
            #pragma unroll
            for (int e = 0; e < 8; ++e) accU[h][mt][e] = 0.f;

    #pragma unroll
    for (int kc = 0; kc < 4; ++kc) {
        #pragma unroll
        for (int half = 0; half < 2; ++half) {
            int nt = w + half * 8;
            v16h bf;
            const _Float16* bp = &sW[(nt * 16 + nl) * RANK + kc * 32];
            #pragma unroll
            for (int e = 0; e < 16; ++e) bf[e] = bp[(e < 8) ? (8 * hi + e) : (16 + 8 * hi + e - 8)];
            #pragma unroll
            for (int mt = 0; mt < 4; ++mt) {
                v16h af;
                const _Float16* ap = &sLow[(mt * 16 + nl) * RANK + kc * 32];
                #pragma unroll
                for (int e = 0; e < 8; ++e) af[e]     = ap[8 * hi + e];
                #pragma unroll
                for (int e = 0; e < 8; ++e) af[8 + e] = ap[16 + 8 * hi + e];
                accU[half][mt] = __builtin_amdgcn_wmma_f32_16x16x32_f16(
                    false, af, false, bf, (short)0, accU[half][mt], false, false);
            }
        }
    }

    // ---------------- gate * residual fused store ---------------------------
    #pragma unroll
    for (int mt = 0; mt < 4; ++mt)
        #pragma unroll
        for (int r = 0; r < 8; ++r) {
            int rg = row0 + mt * 16 + r + 8 * hi;
            float gs = resScale * gate[(size_t)rg * NB + blk];
            #pragma unroll
            for (int half = 0; half < 2; ++half) {
                int col = blk * BS + (w + half * 8) * 16 + nl;
                size_t off = (size_t)rg * H + col;
                out[off] = x[off] + gs * accU[half][mt][r];
            }
        }
}

// ---------------------------------------------------------------------------
extern "C" void kernel_launch(void* const* d_in, const int* in_sizes, int n_in,
                              void* d_out, int out_size, void* d_ws, size_t ws_size,
                              hipStream_t stream) {
    const float* x     = (const float*)d_in[0];
    const float* prior = (const float*)d_in[1];
    const float* sw    = (const float*)d_in[2];
    const float* sb    = (const float*)d_in[3];
    const float* dw    = (const float*)d_in[4];
    const float* uw    = (const float*)d_in[5];
    const float* rps   = (const float*)d_in[6];
    const float* res   = (const float*)d_in[7];
    float* out = (float*)d_out;

    const int rows = in_sizes[0] / H;                 // 8192
    const size_t gate_bytes = (size_t)rows * NB * sizeof(float);
    const size_t wtab_elems = (size_t)NB * BS * RANK; // 524288

    float*    gate = (float*)d_ws;
    _Float16* wdt  = (_Float16*)((char*)d_ws + ((gate_bytes + 255) & ~(size_t)255));
    _Float16* wut  = wdt + wtab_elems;

    SparseSCA_prep_kernel<<<(int)(wtab_elems / 256), 256, 0, stream>>>(dw, uw, wdt, wut);

    SparseSCA_gate_kernel<<<(rows + 7) / 8, 256, 0, stream>>>(
        x, prior, sw, sb, rps, gate, rows);

    dim3 grid(rows / ROWS_TILE, NB);
    SparseSCA_adapter_kernel<<<grid, 256, 0, stream>>>(
        x, wdt, wut, gate, res, out, rows);
}
